// GatedCrossModalAttentionUnit_86912958202445
// MI455X (gfx1250) — matrix operand order repaired and checked by reading, not compiled, not run on hardware
//
#include <hip/hip_runtime.h>

// ---------------------------------------------------------------------------
// GatedCrossModalAttentionUnit for MI455X (gfx1250): bf16 WMMA GEMM pipeline.
// B=16384, QD=2048, KD=1024, HID=1024, NH=16, HD=64.
// ---------------------------------------------------------------------------

#define B_    16384
#define QD_   2048
#define KD_   1024
#define HID_  1024

typedef __attribute__((ext_vector_type(16))) __bf16 v16bf;
typedef __attribute__((ext_vector_type(8)))  __bf16 v8bf;
typedef __attribute__((ext_vector_type(4)))  __bf16 v4bf;
typedef __attribute__((ext_vector_type(8)))  float  v8f;

#define TM 128      // block tile M
#define TN 256      // block tile N
#define TK 32       // K step
#define LDS_S 40    // padded row stride (elements) -> conflict-free b128 LDS reads

__device__ __forceinline__ v16bf bf_cat(const __bf16* p0, const __bf16* p1) {
  v8bf lo = *(const v8bf*)p0;
  v8bf hi = *(const v8bf*)p1;
  v16bf r;
#pragma unroll
  for (int i = 0; i < 8; ++i) { r[i] = lo[i]; r[i + 8] = hi[i]; }
  return r;
}

// ---------------------------------------------------------------------------
// f32 -> bf16 conversion, vectorized x4 (all sizes are multiples of 4)
// ---------------------------------------------------------------------------
__global__ void cvt_bf16_kernel(const float4* __restrict__ src,
                                v4bf* __restrict__ dst, long n4) {
  long i = (long)blockIdx.x * blockDim.x + threadIdx.x;
  const long stride = (long)gridDim.x * blockDim.x;
  for (; i < n4; i += stride) {
    const float4 f = src[i];
    v4bf o;
    o[0] = (__bf16)f.x; o[1] = (__bf16)f.y;
    o[2] = (__bf16)f.z; o[3] = (__bf16)f.w;
    dst[i] = o;
  }
}

// ---------------------------------------------------------------------------
// WMMA GEMM:  D[M,N](f32) = act( A[M,K1]·W1[N,K1]^T (+ A2[M,K2]·W2[N,K2]^T) + bias )
// A, W in bf16 row-major.  Block: 128x256 tile, 8 waves (2Mx4N), wave = 64x64
// => 16 v_wmma_f32_16x16x32_bf16 per K-step per wave.
// ACT: 0 = none, 1 = exact GELU.
// ---------------------------------------------------------------------------
template <int ACT>
__global__ __launch_bounds__(256)
void wmma_gemm_kernel(const __bf16* __restrict__ A1, int lda1, int T1,
                      const __bf16* __restrict__ A2, int lda2, int T2,
                      const __bf16* __restrict__ W1, const __bf16* __restrict__ W2,
                      int ldw, const float* __restrict__ bias,
                      float* __restrict__ D, int N) {
  __shared__ __align__(16) __bf16 sA[2][TM * LDS_S];   // 2 x 10 KB
  __shared__ __align__(16) __bf16 sB[2][TN * LDS_S];   // 2 x 20 KB

  const int tid  = threadIdx.x;
  const int lane = tid & 31;
  const int wid  = tid >> 5;
  const int mw   = wid & 1;          // wave grid 2(M) x 4(N)
  const int nw   = wid >> 1;
  const int half = lane >> 4;
  const int l16  = lane & 15;
  const int gm0  = blockIdx.y * TM;
  const int gn0  = blockIdx.x * TN;

  // global<->LDS staging: 16B chunks; A = 2/thread, B = 4/thread
  const int rr = tid >> 2;           // 0..63
  const int c0 = (tid & 3) * 8;      // 0/8/16/24

  v8f acc[4][4];
  const v8f vzero = {0.f, 0.f, 0.f, 0.f, 0.f, 0.f, 0.f, 0.f};
#pragma unroll
  for (int mi = 0; mi < 4; ++mi)
#pragma unroll
    for (int ni = 0; ni < 4; ++ni) acc[mi][ni] = vzero;

  const int T = T1 + T2;
  uint4 ra[2], rb[4];

  { // prologue: tile 0 -> LDS[0]
    const __bf16* Ab = A1; int la = lda1; const __bf16* Wb = W1;
    if (T1 == 0) { Ab = A2; la = lda2; Wb = W2; }
#pragma unroll
    for (int i = 0; i < 2; ++i)
      ra[i] = *(const uint4*)&Ab[(size_t)(gm0 + rr + 64 * i) * la + c0];
#pragma unroll
    for (int i = 0; i < 4; ++i)
      rb[i] = *(const uint4*)&Wb[(size_t)(gn0 + rr + 64 * i) * ldw + c0];
  }
#pragma unroll
  for (int i = 0; i < 2; ++i)
    *(uint4*)&sA[0][(rr + 64 * i) * LDS_S + c0] = ra[i];
#pragma unroll
  for (int i = 0; i < 4; ++i)
    *(uint4*)&sB[0][(rr + 64 * i) * LDS_S + c0] = rb[i];
  __syncthreads();

  for (int t = 0; t < T; ++t) {
    const int cur = t & 1;

    if (t + 2 < T) {  // L2 prefetch two tiles ahead (global_prefetch_b8)
      const int tp = t + 2;
      const __bf16* Ab; int la; const __bf16* Wb; int k0;
      if (tp < T1) { Ab = A1; la = lda1; Wb = W1; k0 = tp * TK; }
      else         { Ab = A2; la = lda2; Wb = W2; k0 = (tp - T1) * TK; }
      __builtin_prefetch(&Ab[(size_t)(gm0 + rr) * la + k0 + c0], 0, 3);
      __builtin_prefetch(&Wb[(size_t)(gn0 + rr) * ldw + k0 + c0], 0, 3);
    }

    if (t + 1 < T) {  // prefetch next tile into registers
      const int tn = t + 1;
      const __bf16* Ab; int la; const __bf16* Wb; int k0;
      if (tn < T1) { Ab = A1; la = lda1; Wb = W1; k0 = tn * TK; }
      else         { Ab = A2; la = lda2; Wb = W2; k0 = (tn - T1) * TK; }
#pragma unroll
      for (int i = 0; i < 2; ++i)
        ra[i] = *(const uint4*)&Ab[(size_t)(gm0 + rr + 64 * i) * la + k0 + c0];
#pragma unroll
      for (int i = 0; i < 4; ++i)
        rb[i] = *(const uint4*)&Wb[(size_t)(gn0 + rr + 64 * i) * ldw + k0 + c0];
    }

    // compute from LDS[cur]: 16 WMMAs
    v16bf af[4];
#pragma unroll
    for (int mi = 0; mi < 4; ++mi) {
      const int row = mw * 64 + mi * 16 + l16;
      const __bf16* p = &sA[cur][row * LDS_S + half * 8];
      af[mi] = bf_cat(p, p + 16);   // K-chunks {kb..kb+7, kb+16..kb+23}
    }
#pragma unroll
    for (int ni = 0; ni < 4; ++ni) {
      const int n = nw * 64 + ni * 16 + l16;
      const __bf16* p = &sB[cur][n * LDS_S + half * 8];
      const v16bf bfrag = bf_cat(p, p + 16);  // K-chunks {kb..kb+7, kb+16..kb+23}
#pragma unroll
      for (int mi = 0; mi < 4; ++mi) {
        acc[mi][ni] = __builtin_amdgcn_wmma_f32_16x16x32_bf16(
            false, af[mi], false, bfrag, (short)0, acc[mi][ni], false, false);
      }
    }
    __syncthreads();
    if (t + 1 < T) {
      const int nxt = cur ^ 1;
#pragma unroll
      for (int i = 0; i < 2; ++i)
        *(uint4*)&sA[nxt][(rr + 64 * i) * LDS_S + c0] = ra[i];
#pragma unroll
      for (int i = 0; i < 4; ++i)
        *(uint4*)&sB[nxt][(rr + 64 * i) * LDS_S + c0] = rb[i];
      __syncthreads();
    }
  }

  // epilogue: bias (+ GELU), f32 store
#pragma unroll
  for (int mi = 0; mi < 4; ++mi) {
#pragma unroll
    for (int ni = 0; ni < 4; ++ni) {
#pragma unroll
      for (int i = 0; i < 8; ++i) {
        const int gm = gm0 + mw * 64 + mi * 16 + half * 8 + i;
        const int gn = gn0 + nw * 64 + ni * 16 + l16;
        float x = acc[mi][ni][i] + bias[gn];
        if (ACT == 1) x = 0.5f * x * (1.f + erff(x * 0.70710678118654752f));
        D[(size_t)gm * N + gn] = x;
      }
    }
  }
}

// ---------------------------------------------------------------------------
// Row LayerNorm (in-place safe): one 256-thread block per row.
// ---------------------------------------------------------------------------
__global__ __launch_bounds__(256)
void ln_kernel(const float* __restrict__ x, const float* __restrict__ w,
               const float* __restrict__ b, float* __restrict__ y, int D) {
  __shared__ float rs[8], rss[8];
  const int row = blockIdx.x;
  const float* xr = x + (size_t)row * D;
  float s = 0.f, ss = 0.f;
  for (int j = threadIdx.x; j < D; j += 256) {
    const float v = xr[j];
    s += v; ss += v * v;
  }
#pragma unroll
  for (int d = 16; d >= 1; d >>= 1) {
    s  += __shfl_xor(s, d, 32);
    ss += __shfl_xor(ss, d, 32);
  }
  if ((threadIdx.x & 31) == 0) { rs[threadIdx.x >> 5] = s; rss[threadIdx.x >> 5] = ss; }
  __syncthreads();
  float S = 0.f, SS = 0.f;
#pragma unroll
  for (int i = 0; i < 8; ++i) { S += rs[i]; SS += rss[i]; }
  const float mean = S / D;
  const float var  = SS / D - mean * mean;
  const float inv  = rsqrtf(var + 1e-5f);
  float* yr = y + (size_t)row * D;
  for (int j = threadIdx.x; j < D; j += 256)
    yr[j] = (xr[j] - mean) * inv * w[j] + b[j];
}

// ---------------------------------------------------------------------------
// Per-sample head-mix attention: one wave per sample.
//   scores = q[16,64]·k[16,64]^T / 8 ; softmax ; attended = attn·v[16,64]
// Writes attended as bf16 (feeds Wo GEMM directly).
// ---------------------------------------------------------------------------
__global__ __launch_bounds__(256)
void attn_kernel(const float* __restrict__ q, const float* __restrict__ k,
                 const float* __restrict__ v, __bf16* __restrict__ att) {
  __shared__ float sAttn[8][256];
  const int tid = threadIdx.x, lane = tid & 31, wid = tid >> 5;
  const int half = lane >> 4, l16 = lane & 15;
  const size_t s = (size_t)blockIdx.x * 8 + wid;
  const float* qr = q + s * (size_t)HID_;
  const float* kr = k + s * (size_t)HID_;
  const float* vr = v + s * (size_t)HID_;

  // ---- scores = q·k^T (16x16, K=64 => two 16x16x32 WMMAs) ----
  v8f sc = {0.f, 0.f, 0.f, 0.f, 0.f, 0.f, 0.f, 0.f};
#pragma unroll
  for (int c = 0; c < 2; ++c) {
    v16bf a, bm;
    const int kb = c * 32 + half * 8;
#pragma unroll
    for (int e = 0; e < 8; ++e) {
      a[e]     = (__bf16)qr[l16 * 64 + kb + e];
      a[e + 8] = (__bf16)qr[l16 * 64 + kb + 16 + e];
    }
    const int db = c * 32 + half * 8;
#pragma unroll
    for (int e = 0; e < 16; ++e) bm[e] = (__bf16)kr[l16 * 64 + db + (e < 8 ? e : e + 8)];
    sc = __builtin_amdgcn_wmma_f32_16x16x32_bf16(false, a, false, bm,
                                                 (short)0, sc, false, false);
  }

  // ---- softmax over j (columns; 16 lanes within each half-wave) ----
  float p[8];
#pragma unroll
  for (int i = 0; i < 8; ++i) p[i] = sc[i] * 0.125f;
#pragma unroll
  for (int i = 0; i < 8; ++i) {
    float mx = p[i];
#pragma unroll
    for (int d = 8; d >= 1; d >>= 1) mx = fmaxf(mx, __shfl_xor(mx, d, 32));
    float e = __expf(p[i] - mx);
    float sm = e;
#pragma unroll
    for (int d = 8; d >= 1; d >>= 1) sm += __shfl_xor(sm, d, 32);
    p[i] = e / sm;
  }

  // ---- relayout attn (C-layout -> A-fragment) via LDS ----
#pragma unroll
  for (int i = 0; i < 8; ++i)
    sAttn[wid][(i + half * 8) * 16 + l16] = p[i];
  __syncthreads();

  v16bf aa;
#pragma unroll
  for (int e = 0; e < 8; ++e) {
    aa[e]     = (__bf16)sAttn[wid][l16 * 16 + half * 8 + e];  // K = 0..15
    aa[e + 8] = (__bf16)0.f;                                  // K = 16..31 pad
  }
  __syncthreads();

  // ---- attended = attn·v  (4 WMMAs across d tiles of 16) ----
#pragma unroll
  for (int ni = 0; ni < 4; ++ni) {
    v16bf bb;
#pragma unroll
    for (int e = 0; e < 16; ++e) {
      const int kk = (e < 8) ? half * 8 + e : 16 + half * 8 + (e - 8);
      bb[e] = (kk < 16) ? (__bf16)vr[kk * 64 + ni * 16 + l16] : (__bf16)0.f;
    }
    v8f ac = {0.f, 0.f, 0.f, 0.f, 0.f, 0.f, 0.f, 0.f};
    ac = __builtin_amdgcn_wmma_f32_16x16x32_bf16(false, aa, false, bb,
                                                 (short)0, ac, false, false);
#pragma unroll
    for (int i = 0; i < 8; ++i) {
      const int h = half * 8 + i;
      att[s * (size_t)HID_ + h * 64 + ni * 16 + l16] = (__bf16)ac[i];
    }
  }
}

// ---------------------------------------------------------------------------
// gate = sigmoid(X[row,:]·w + b) : one wave per row
// ---------------------------------------------------------------------------
__global__ __launch_bounds__(256)
void gemv_sigmoid_kernel(const float* __restrict__ X, const float* __restrict__ w,
                         const float* __restrict__ b0, float* __restrict__ out, int D) {
  const int lane = threadIdx.x & 31, wid = threadIdx.x >> 5;
  const size_t row = (size_t)blockIdx.x * 8 + wid;
  const float* xr = X + row * D;
  float s = 0.f;
  for (int j = lane; j < D; j += 32) s += xr[j] * w[j];
#pragma unroll
  for (int d = 16; d >= 1; d >>= 1) s += __shfl_xor(s, d, 32);
  if (lane == 0) out[row] = 1.f / (1.f + __expf(-(s + b0[0])));
}

// ---------------------------------------------------------------------------
// out = query + gate*chem*ln_out
// ---------------------------------------------------------------------------
__global__ void final_kernel(const float* __restrict__ qf, const float* __restrict__ lno,
                             const float* __restrict__ gate, const float* __restrict__ chem,
                             float* __restrict__ out, long n) {
  long i = (long)blockIdx.x * blockDim.x + threadIdx.x;
  const long stride = (long)gridDim.x * blockDim.x;
  for (; i < n; i += stride) {
    const long b = i >> 11;  // / QD_ (2048)
    out[i] = qf[i] + gate[b] * chem[b] * lno[i];
  }
}

// ---------------------------------------------------------------------------
extern "C" void kernel_launch(void* const* d_in, const int* in_sizes, int n_in,
                              void* d_out, int out_size, void* d_ws, size_t ws_size,
                              hipStream_t stream) {
  (void)in_sizes; (void)n_in; (void)out_size; (void)ws_size;
  const float* qf   = (const float*)d_in[0];
  const float* kf   = (const float*)d_in[1];
  const float* Wq   = (const float*)d_in[2];
  const float* bq   = (const float*)d_in[3];
  const float* lnqw = (const float*)d_in[4];
  const float* lnqb = (const float*)d_in[5];
  const float* Wk   = (const float*)d_in[6];
  const float* bk   = (const float*)d_in[7];
  const float* lnkw = (const float*)d_in[8];
  const float* lnkb = (const float*)d_in[9];
  /* d_in[10] Wvq, d_in[11] bvq: dead code in reference (v_q unused) */
  const float* Wvk  = (const float*)d_in[12];
  const float* bvk  = (const float*)d_in[13];
  const float* Wo   = (const float*)d_in[14];
  const float* bo   = (const float*)d_in[15];
  const float* lnow = (const float*)d_in[16];
  const float* lnob = (const float*)d_in[17];
  const float* Wg1  = (const float*)d_in[18];
  const float* bg1  = (const float*)d_in[19];
  const float* Wg2  = (const float*)d_in[20];
  const float* bg2  = (const float*)d_in[21];
  const float* Wc1  = (const float*)d_in[22];
  const float* bc1  = (const float*)d_in[23];
  const float* Wc2  = (const float*)d_in[24];
  const float* bc2  = (const float*)d_in[25];
  float* out = (float*)d_out;

  // ---- workspace arena ----
  char* base = (char*)d_ws;
  size_t off = 0;
  auto alloc = [&](size_t bytes) -> char* {
    char* p = base + off;
    off = (off + bytes + 255) & ~(size_t)255;
    return p;
  };
  __bf16* qbf    = (__bf16*)alloc((size_t)B_ * QD_ * 2);
  __bf16* kbf    = (__bf16*)alloc((size_t)B_ * KD_ * 2);
  __bf16* Wq_b   = (__bf16*)alloc((size_t)HID_ * QD_ * 2);
  __bf16* Wk_b   = (__bf16*)alloc((size_t)HID_ * KD_ * 2);
  __bf16* Wvk_b  = (__bf16*)alloc((size_t)HID_ * KD_ * 2);
  __bf16* Wo_b   = (__bf16*)alloc((size_t)QD_ * HID_ * 2);
  __bf16* Wg1_b  = (__bf16*)alloc((size_t)HID_ * (QD_ + KD_) * 2);
  __bf16* Wc1_b  = (__bf16*)alloc((size_t)HID_ * (QD_ + KD_) * 2);
  float*  q32    = (float*)alloc((size_t)B_ * HID_ * 4);
  float*  k32    = (float*)alloc((size_t)B_ * HID_ * 4);
  float*  vk32   = (float*)alloc((size_t)B_ * HID_ * 4);
  __bf16* att_b  = (__bf16*)alloc((size_t)B_ * HID_ * 2);
  float*  outp   = (float*)alloc((size_t)B_ * QD_ * 4);
  float*  hbuf   = (float*)alloc((size_t)B_ * HID_ * 4);  // g1 then c1 (reused)
  float*  gateb  = (float*)alloc((size_t)B_ * 4);
  float*  chemb  = (float*)alloc((size_t)B_ * 4);

  // ---- 1. f32 -> bf16 conversions (x4 vectorized) ----
  auto cvt = [&](const float* s, __bf16* d, long n) {
    cvt_bf16_kernel<<<2048, 256, 0, stream>>>((const float4*)s, (v4bf*)d, n / 4);
  };
  cvt(qf,  qbf,  (long)B_ * QD_);
  cvt(kf,  kbf,  (long)B_ * KD_);
  cvt(Wq,  Wq_b, (long)HID_ * QD_);
  cvt(Wk,  Wk_b, (long)HID_ * KD_);
  cvt(Wvk, Wvk_b,(long)HID_ * KD_);
  cvt(Wo,  Wo_b, (long)QD_ * HID_);
  cvt(Wg1, Wg1_b,(long)HID_ * (QD_ + KD_));
  cvt(Wc1, Wc1_b,(long)HID_ * (QD_ + KD_));

  const dim3 blk(256);
  const dim3 grid_h(HID_ / TN, B_ / TM);   // N=1024 -> (4,128)
  const dim3 grid_o(QD_ / TN,  B_ / TM);   // N=2048 -> (8,128)

  // ---- 2. projections ----
  wmma_gemm_kernel<0><<<grid_h, blk, 0, stream>>>(
      qbf, QD_, QD_ / TK, nullptr, 0, 0, Wq_b, nullptr, QD_, bq, q32, HID_);
  wmma_gemm_kernel<0><<<grid_h, blk, 0, stream>>>(
      kbf, KD_, KD_ / TK, nullptr, 0, 0, Wk_b, nullptr, KD_, bk, k32, HID_);
  wmma_gemm_kernel<0><<<grid_h, blk, 0, stream>>>(
      kbf, KD_, KD_ / TK, nullptr, 0, 0, Wvk_b, nullptr, KD_, bvk, vk32, HID_);

  // ---- 3. LayerNorm q, k (in place) ----
  ln_kernel<<<B_, blk, 0, stream>>>(q32, lnqw, lnqb, q32, HID_);
  ln_kernel<<<B_, blk, 0, stream>>>(k32, lnkw, lnkb, k32, HID_);

  // ---- 4. per-sample attention -> attended (bf16) ----
  attn_kernel<<<B_ / 8, blk, 0, stream>>>(q32, k32, vk32, att_b);

  // ---- 5. output projection + LayerNorm ----
  wmma_gemm_kernel<0><<<grid_o, blk, 0, stream>>>(
      att_b, HID_, HID_ / TK, nullptr, 0, 0, Wo_b, nullptr, HID_, bo, outp, QD_);
  ln_kernel<<<B_, blk, 0, stream>>>(outp, lnow, lnob, outp, QD_);

  // ---- 6. gate path: gelu(concat(q,k)·Wg1^T + bg1) via split-K fused GEMM ----
  wmma_gemm_kernel<1><<<grid_h, blk, 0, stream>>>(
      qbf, QD_, QD_ / TK, kbf, KD_, KD_ / TK,
      Wg1_b, Wg1_b + QD_, QD_ + KD_, bg1, hbuf, HID_);
  gemv_sigmoid_kernel<<<B_ / 8, blk, 0, stream>>>(hbuf, Wg2, bg2, gateb, HID_);

  // ---- 7. chem path (reuse hbuf) ----
  wmma_gemm_kernel<1><<<grid_h, blk, 0, stream>>>(
      qbf, QD_, QD_ / TK, kbf, KD_, KD_ / TK,
      Wc1_b, Wc1_b + QD_, QD_ + KD_, bc1, hbuf, HID_);
  gemv_sigmoid_kernel<<<B_ / 8, blk, 0, stream>>>(hbuf, Wc2, bc2, chemb, HID_);

  // ---- 8. gated residual ----
  final_kernel<<<4096, blk, 0, stream>>>(qf, outp, gateb, chemb, out,
                                         (long)B_ * QD_);
}
